// ConvSplitAttn_49065706390044
// MI455X (gfx1250) — matrix operand order repaired and checked by reading, not compiled, not run on hardware
//
#include <hip/hip_runtime.h>

// ---------------------------------------------------------------------------
// Types for CDNA5 WMMA
// ---------------------------------------------------------------------------
typedef __attribute__((ext_vector_type(16))) __bf16          v16bf;
typedef __attribute__((ext_vector_type(16))) unsigned short  v16u16;
typedef __attribute__((ext_vector_type(8)))  float           v8f;
typedef int v4i_ __attribute__((vector_size(16)));           // b128 payload type

// ---- optional gfx1250 async global->LDS path (compile-safe feature gate) ----
#if defined(__has_builtin)
#if __has_builtin(__builtin_amdgcn_global_load_async_to_lds_b128) && \
    __has_builtin(__builtin_amdgcn_s_wait_asynccnt)
#define HAVE_ASYNC_LDS 1
#endif
#endif
#ifndef HAVE_ASYNC_LDS
#define HAVE_ASYNC_LDS 0
#endif

__device__ __forceinline__ void async_b128(const unsigned short* g, unsigned short* l) {
#if HAVE_ASYNC_LDS
    __builtin_amdgcn_global_load_async_to_lds_b128(
        (__attribute__((address_space(1))) v4i_*)(unsigned long long)g,
        (__attribute__((address_space(3))) v4i_*)(unsigned int)(unsigned long long)l,
        0, 0);
#else
    (void)g; (void)l;
#endif
}
__device__ __forceinline__ void wait_async_publish() {
#if HAVE_ASYNC_LDS
    __builtin_amdgcn_s_wait_asynccnt(0);
#endif
    __syncthreads();
}

__device__ __forceinline__ unsigned short f32_to_bf16(float f) {
    unsigned int u = __float_as_uint(f);
    u += 0x7FFFu + ((u >> 16) & 1u);           // round-to-nearest-even
    return (unsigned short)(u >> 16);
}
__device__ __forceinline__ float bf16_to_f32(unsigned int h) {
    return __uint_as_float(h << 16);
}

// ---------------------------------------------------------------------------
// dims: B=16,N=32,D=64,H=8,S=8,LT=128,DE=256,L=16,G=8
// samples=512, head-batches=128, rows=N*S=256, DL=1024
//
// Padded activation rows for k=3 conv im2col:
//   row stride PLT=144 elems, data at [POFF+t], zero sentinels at [7],[136]
// ---------------------------------------------------------------------------
constexpr int PLT  = 144;
constexpr int POFF = 8;
constexpr unsigned KADV96 = 32u * PLT;   // 96 K == 32 rows == 4608 elems

constexpr int B_TRANS  = 0;   // B[k][n] = src[n*ldb + k]
constexpr int B_IM2COL = 1;   // B[ci*3+dk][t] = src[ci*PLT + POFF + t+dk-1]

constexpr int E_QKV   = 0;
constexpr int E_ATTN  = 1;
constexpr int E_AV    = 2;
constexpr int E_MERGE = 3;
constexpr int E_FF1   = 4;
constexpr int E_FF2   = 5;

struct GemmP {
    const unsigned short* A;  size_t strideA; int lda;
    const unsigned short* B;  size_t strideB; int ldb;
    int K;
    const float* bias;
    const float* resid;
    const float* qn;
    const float* kn;
    unsigned short* outb;
    float*          outf;
};

// ---------------------------------------------------------------------------
// WMMA GEMM: C(MxN) = A(MxK,row,bf16) * B(KxN,bf16), f32 accum
// 64(M) x 128(N) block tile, BK=32, 256 threads = 8 waves (wave = 32M x 32N)
// 4 WMMA per wave per K-step, double-buffered LDS, async global->LDS staging
// ---------------------------------------------------------------------------
template<int BMODE, int EPI>
__global__ __launch_bounds__(256)
void gemm_wmma_k(GemmP p)
{
    __shared__ __align__(16) unsigned short AsB[2][64][40];
    __shared__ __align__(16) unsigned short BsB[2][128][40];

    const int tid  = threadIdx.x;
    const int lane = tid & 31;
    const int w    = tid >> 5;
    const int wr   = w >> 2;          // 0..1  (M 32-row half)
    const int wc   = w & 3;           // 0..3  (N 32-col quarter)
    const int lm   = lane & 15;
    const int half = lane >> 4;

    const int z  = blockIdx.z;
    const int m0 = blockIdx.y * 64;
    const int n0 = blockIdx.x * 128;

    const unsigned short* __restrict__ Bb = p.B + (size_t)z * p.strideB;

    // staging coordinates
    const int ar  = tid >> 2;                  // A: 4 threads x 8 elems per row
    const int ac8 = (tid & 3) * 8;
    const unsigned short* Aptr = p.A + (size_t)z * p.strideA
                               + (size_t)(m0 + ar) * p.lda + ac8;
    const int bn  = (BMODE == B_TRANS) ? (tid >> 2) : (tid & 63);   // second col at bn+64
    const int bk8 = (BMODE == B_TRANS) ? (tid & 3) * 8 : (tid >> 6) * 8;
    const unsigned short* Bptr0 =
        (BMODE == B_TRANS) ? (Bb + (size_t)(n0 + bn) * p.ldb + bk8) : Bb;
    const unsigned short* Bptr1 =
        (BMODE == B_TRANS) ? (Bptr0 + (size_t)64 * p.ldb) : Bb;

    v8f acc[4];
    #pragma unroll
    for (int i = 0; i < 4; ++i) acc[i] = v8f{0.f,0.f,0.f,0.f,0.f,0.f,0.f,0.f};

    uint4 aReg, bReg0, bReg1;

    auto mmaStep = [&](int buf) {
        const unsigned short (*As)[40]  = AsB[buf];
        const unsigned short (*BsT)[40] = BsB[buf];
        v16u16 bf0, bf1, af0, af1;
        const int nn = wc * 32 + lm;          // B: lane->col, K = 16*half + e
        ((uint4*)&bf0)[0] = *(const uint4*)&BsT[nn][16 * half];
        ((uint4*)&bf0)[1] = *(const uint4*)&BsT[nn][16 * half + 8];
        ((uint4*)&bf1)[0] = *(const uint4*)&BsT[nn + 16][16 * half];
        ((uint4*)&bf1)[1] = *(const uint4*)&BsT[nn + 16][16 * half + 8];
        const int mm = wr * 32 + lm;          // A: lane->row, K = 16*half + e
        ((uint4*)&af0)[0] = *(const uint4*)&As[mm][16 * half];
        ((uint4*)&af0)[1] = *(const uint4*)&As[mm][16 * half + 8];
        ((uint4*)&af1)[0] = *(const uint4*)&As[mm + 16][16 * half];
        ((uint4*)&af1)[1] = *(const uint4*)&As[mm + 16][16 * half + 8];
        const v16bf a0 = __builtin_bit_cast(v16bf, af0);
        const v16bf a1 = __builtin_bit_cast(v16bf, af1);
        const v16bf b0 = __builtin_bit_cast(v16bf, bf0);
        const v16bf b1 = __builtin_bit_cast(v16bf, bf1);
        acc[0] = __builtin_amdgcn_wmma_f32_16x16x32_bf16(false, a0, false, b0,
                                                         (short)0, acc[0], false, false);
        acc[1] = __builtin_amdgcn_wmma_f32_16x16x32_bf16(false, a1, false, b0,
                                                         (short)0, acc[1], false, false);
        acc[2] = __builtin_amdgcn_wmma_f32_16x16x32_bf16(false, a0, false, b1,
                                                         (short)0, acc[2], false, false);
        acc[3] = __builtin_amdgcn_wmma_f32_16x16x32_bf16(false, a1, false, b1,
                                                         (short)0, acc[3], false, false);
    };

    if constexpr (BMODE == B_TRANS) {
#if HAVE_ASYNC_LDS
        // ---- fully async staging: global -> LDS, tracked on ASYNCcnt ----
        auto stageAsync = [&](int kq, int buf) {
            async_b128(Aptr  + kq, &AsB[buf][ar][ac8]);
            async_b128(Bptr0 + kq, &BsB[buf][bn][bk8]);
            async_b128(Bptr1 + kq, &BsB[buf][bn + 64][bk8]);
        };
        stageAsync(0, 0);
        wait_async_publish();
        for (int k0 = 0; k0 < p.K; k0 += 64) {
            stageAsync(k0 + 32, 1);           // in flight during MMA on buf0
            mmaStep(0);
            wait_async_publish();
            const bool more = (k0 + 64 < p.K);
            if (more) stageAsync(k0 + 64, 0);
            mmaStep(1);
            wait_async_publish();
        }
#else
        auto loadT = [&](int kq) {
            aReg  = *(const uint4*)(Aptr  + kq);
            bReg0 = *(const uint4*)(Bptr0 + kq);
            bReg1 = *(const uint4*)(Bptr1 + kq);
        };
        auto storeT = [&](int buf) {
            *(uint4*)&AsB[buf][ar][ac8]      = aReg;
            *(uint4*)&BsB[buf][bn][bk8]      = bReg0;
            *(uint4*)&BsB[buf][bn + 64][bk8] = bReg1;
        };
        loadT(0);
        storeT(0);
        __syncthreads();
        for (int k0 = 0; k0 < p.K; k0 += 64) {
            loadT(k0 + 32);
            mmaStep(0);
            storeT(1);
            __syncthreads();
            const bool more = (k0 + 64 < p.K);
            if (more) loadT(k0 + 64);
            mmaStep(1);
            if (more) storeT(0);
            __syncthreads();
        }
#endif
    } else {
        // ---- phase-periodic gather offsets: 3 phases x 8 slots ----
        unsigned ofs0[8], ofs1[8], ofs2[8];
        {
            const int t = n0 + bn;
            #pragma unroll
            for (int j = 0; j < 8; ++j) {
                #pragma unroll
                for (int s = 0; s < 3; ++s) {
                    const int kk = s * 32 + bk8 + j;
                    const int ci = kk / 3;
                    const int dk = kk - 3 * ci;
                    const unsigned o = (unsigned)(ci * PLT + (POFF - 1) + t + dk);
                    if (s == 0) ofs0[j] = o; else if (s == 1) ofs1[j] = o; else ofs2[j] = o;
                }
            }
        }
        auto loadI = [&](int kq, unsigned (&of)[8], int buf) {
#if HAVE_ASYNC_LDS
            async_b128(Aptr + kq, &AsB[buf][ar][ac8]);   // A tile via async DMA
#else
            aReg = *(const uint4*)(Aptr + kq);
#endif
            (void)buf;
            unsigned int va[8], vb[8];
            #pragma unroll
            for (int j = 0; j < 8; ++j) {      // issue all gathers first (one clause)
                va[j] = Bb[of[j]];
                vb[j] = Bb[of[j] + 64];        // second n-column
            }
            unsigned int w0[4], w1[4];
            #pragma unroll
            for (int jj = 0; jj < 4; ++jj) {
                w0[jj] = va[2 * jj] | (va[2 * jj + 1] << 16);
                w1[jj] = vb[2 * jj] | (vb[2 * jj + 1] << 16);
                of[2 * jj]     += KADV96;      // next use is 96 K later
                of[2 * jj + 1] += KADV96;
            }
            bReg0 = make_uint4(w0[0], w0[1], w0[2], w0[3]);
            bReg1 = make_uint4(w1[0], w1[1], w1[2], w1[3]);
        };
        auto storeI = [&](int buf) {
#if !HAVE_ASYNC_LDS
            *(uint4*)&AsB[buf][ar][ac8] = aReg;
#endif
            *(uint4*)&BsB[buf][bn][bk8]      = bReg0;
            *(uint4*)&BsB[buf][bn + 64][bk8] = bReg1;
        };
        // ---- 6-step pipeline (K/32 multiple of 6) ----
        loadI(0, ofs0, 0);
        storeI(0);
        wait_async_publish();
        for (int k0 = 0; k0 < p.K; k0 += 192) {
            loadI(k0 + 32,  ofs1, 1); mmaStep(0); storeI(1); wait_async_publish();
            loadI(k0 + 64,  ofs2, 0); mmaStep(1); storeI(0); wait_async_publish();
            loadI(k0 + 96,  ofs0, 1); mmaStep(0); storeI(1); wait_async_publish();
            loadI(k0 + 128, ofs1, 0); mmaStep(1); storeI(0); wait_async_publish();
            loadI(k0 + 160, ofs2, 1); mmaStep(0); storeI(1); wait_async_publish();
            const bool more = (k0 + 192 < p.K);
            if (more) loadI(k0 + 192, ofs0, 0);
            mmaStep(1);
            if (more) storeI(0);
            wait_async_publish();
        }
    }

    // ---- epilogue: C lane layout: n = lm ; m = r + 8*half per 16x16 tile ----
    #pragma unroll
    for (int tn = 0; tn < 2; ++tn) {
        const int n = n0 + wc * 32 + tn * 16 + lm;
        #pragma unroll
        for (int tm = 0; tm < 2; ++tm) {
            const int ai = tn * 2 + tm;
            if constexpr (EPI == E_AV) {
                const int mbase = m0 + wr * 32 + tm * 16 + 8 * half;  // mult of 8
                const int b_ = z >> 3, h = z & 7;
                const int nt = mbase >> 3;
                const int cch = n >> 4, l = n & 15;
                const size_t row = ((size_t)(b_ * 32 + nt) * 512 + (h * 64 + cch)) * PLT;
                unsigned int pw[4];
                #pragma unroll
                for (int jj = 0; jj < 4; ++jj)
                    pw[jj] = (unsigned int)f32_to_bf16(acc[ai][2 * jj])
                           | ((unsigned int)f32_to_bf16(acc[ai][2 * jj + 1]) << 16);
                *(uint4*)(p.outb + row + POFF + l * 8) = make_uint4(pw[0], pw[1], pw[2], pw[3]);
                if (l == 0)  p.outb[row + POFF - 1]   = 0;   // sentinels
                if (l == 15) p.outb[row + POFF + 128] = 0;
            } else {
                #pragma unroll
                for (int r = 0; r < 8; ++r) {
                    const int m = m0 + wr * 32 + tm * 16 + 8 * half + r;
                    float cval = acc[ai][r];
                    if constexpr (EPI == E_QKV) {
                        cval += p.bias[m];
                        const int T = m >> 9, rem = m & 511, h = rem >> 6, cch = rem & 63;
                        const int b_ = z >> 5, nt = z & 31;
                        const int l = n >> 3, si = n & 7;
                        size_t dst;
                        if (T < 2)   // q,k : [bh][row][d]
                            dst = (size_t)T * 33554432ull
                                + ((size_t)(b_ * 8 + h) * 256 + (nt * 8 + si)) * 1024
                                + (cch * 16 + l);
                        else         // v transposed : [bh][d][row]
                            dst = 67108864ull
                                + ((size_t)(b_ * 8 + h) * 1024 + (cch * 16 + l)) * 256
                                + (nt * 8 + si);
                        p.outb[dst] = f32_to_bf16(cval);
                    } else if constexpr (EPI == E_ATTN) {
                        const float sq = p.qn[z * 256 + m] + p.kn[z * 256 + n] - 2.f * cval;
                        const float at = rsqrtf(fmaxf(sq, 1e-12f));
                        p.outb[(size_t)z * 65536 + m * 256 + n] = f32_to_bf16(at);
                    } else if constexpr (EPI == E_MERGE) {
                        const size_t idx = (size_t)z * 8192 + m * 128 + n;
                        p.outf[idx] = cval + p.bias[m] + p.resid[idx];
                    } else if constexpr (EPI == E_FF1) {
                        float hv = cval + p.bias[m];
                        hv = hv / (1.f + __expf(-hv));   // swish
                        const size_t row = (size_t)z * 36864 + m * PLT;
                        p.outb[row + POFF + n] = f32_to_bf16(hv);
                        if (n == 0)   p.outb[row + POFF - 1]   = 0;
                        if (n == 127) p.outb[row + POFF + 128] = 0;
                    } else { // E_FF2
                        const size_t idx = (size_t)z * 8192 + m * 128 + n;
                        p.outf[idx] = cval + p.bias[m] + p.resid[idx];
                    }
                }
            }
        }
    }
}

// ---------------------------------------------------------------------------
// GroupNorm (8 groups over (64,128)) : f32 in -> bf16 padded rows (stride PLT)
// ---------------------------------------------------------------------------
__global__ __launch_bounds__(256)
void groupnorm_k(const float* __restrict__ x, const float* __restrict__ g,
                 const float* __restrict__ bt, unsigned short* __restrict__ out)
{
    const int blk = blockIdx.x, smp = blk >> 3, grp = blk & 7;
    const float* base = x + ((size_t)smp * 64 + grp * 8) * 128;
    unsigned short* ob = out + ((size_t)smp * 64 + grp * 8) * PLT;

    const float4 v4 = *(const float4*)(base + threadIdx.x * 4);
    float s1 = v4.x + v4.y + v4.z + v4.w;
    float s2 = v4.x * v4.x + v4.y * v4.y + v4.z * v4.z + v4.w * v4.w;

    __shared__ float a[256], b[256];
    a[threadIdx.x] = s1; b[threadIdx.x] = s2;
    __syncthreads();
    for (int st = 128; st > 0; st >>= 1) {
        if (threadIdx.x < st) { a[threadIdx.x] += a[threadIdx.x + st];
                                b[threadIdx.x] += b[threadIdx.x + st]; }
        __syncthreads();
    }
    const float mean = a[0] * (1.f / 1024.f);
    const float var  = b[0] * (1.f / 1024.f) - mean * mean;
    const float rs   = rsqrtf(var + 1e-5f);

    const int row = threadIdx.x >> 5;
    const int ch  = grp * 8 + row;
    const float ga = g[ch], be = bt[ch];
    const float r0 = (v4.x - mean) * rs * ga + be;
    const float r1 = (v4.y - mean) * rs * ga + be;
    const float r2 = (v4.z - mean) * rs * ga + be;
    const float r3 = (v4.w - mean) * rs * ga + be;
    const unsigned int o0 = (unsigned int)f32_to_bf16(r0) | ((unsigned int)f32_to_bf16(r1) << 16);
    const unsigned int o1 = (unsigned int)f32_to_bf16(r2) | ((unsigned int)f32_to_bf16(r3) << 16);
    unsigned short* orow = ob + row * PLT;
    *(uint2*)(orow + POFF + (threadIdx.x & 31) * 4) = make_uint2(o0, o1);
    if ((threadIdx.x & 31) == 0) {
        orow[POFF - 1]   = 0;
        orow[POFF + 128] = 0;
    }
}

// row sum-of-squares of bf16 rows of length 1024 (for cdist)
__global__ __launch_bounds__(256)
void rownorm_k(const unsigned short* __restrict__ src, float* __restrict__ dst)
{
    const unsigned short* p = src + (size_t)blockIdx.x * 1024;
    const uint2 u = *(const uint2*)(p + threadIdx.x * 4);
    const float v0 = bf16_to_f32(u.x & 0xffffu), v1 = bf16_to_f32(u.x >> 16);
    const float v2 = bf16_to_f32(u.y & 0xffffu), v3 = bf16_to_f32(u.y >> 16);
    float s = v0 * v0 + v1 * v1 + v2 * v2 + v3 * v3;

    __shared__ float sh[256];
    sh[threadIdx.x] = s; __syncthreads();
    for (int st = 128; st > 0; st >>= 1) {
        if (threadIdx.x < st) sh[threadIdx.x] += sh[threadIdx.x + st];
        __syncthreads();
    }
    if (threadIdx.x == 0) dst[blockIdx.x] = sh[0];
}

// f32 -> bf16 pack, 8 elements per thread (n multiple of 8)
__global__ __launch_bounds__(256)
void cvt_bf16_k(const float* __restrict__ src, unsigned short* __restrict__ dst, int n8)
{
    const int i = blockIdx.x * 256 + threadIdx.x;
    if (i < n8) {
        const float4 a = ((const float4*)src)[2 * i];
        const float4 b = ((const float4*)src)[2 * i + 1];
        const unsigned int w0 = (unsigned int)f32_to_bf16(a.x) | ((unsigned int)f32_to_bf16(a.y) << 16);
        const unsigned int w1 = (unsigned int)f32_to_bf16(a.z) | ((unsigned int)f32_to_bf16(a.w) << 16);
        const unsigned int w2 = (unsigned int)f32_to_bf16(b.x) | ((unsigned int)f32_to_bf16(b.y) << 16);
        const unsigned int w3 = (unsigned int)f32_to_bf16(b.z) | ((unsigned int)f32_to_bf16(b.w) << 16);
        ((uint4*)dst)[i] = make_uint4(w0, w1, w2, w3);
    }
}

// ---------------------------------------------------------------------------
// Host orchestration
// ---------------------------------------------------------------------------
extern "C" void kernel_launch(void* const* d_in, const int* in_sizes, int n_in,
                              void* d_out, int out_size, void* d_ws, size_t ws_size,
                              hipStream_t stream)
{
    const float* x     = (const float*)d_in[0];
    const float* gn1_g = (const float*)d_in[1];
    const float* gn1_b = (const float*)d_in[2];
    const float* w_qkv = (const float*)d_in[3];
    const float* b_qkv = (const float*)d_in[4];
    const float* w_mrg = (const float*)d_in[5];
    const float* b_mrg = (const float*)d_in[6];
    const float* gn2_g = (const float*)d_in[7];
    const float* gn2_b = (const float*)d_in[8];
    const float* w_ff1 = (const float*)d_in[9];
    const float* b_ff1 = (const float*)d_in[10];
    const float* w_ff2 = (const float*)d_in[11];
    const float* b_ff2 = (const float*)d_in[12];
    float* out = (float*)d_out;
    (void)in_sizes; (void)n_in; (void)out_size; (void)ws_size;

    char* base = (char*)d_ws;
    size_t off = 0;
    auto carve = [&](size_t bytes) -> void* {
        void* p = base + off;
        off = (off + bytes + 255) & ~(size_t)255;
        return p;
    };

    unsigned short* xn   = (unsigned short*)carve(512ull*64*PLT * 2);   // GN1(x), padded
    unsigned short* wq   = (unsigned short*)carve(294912ull  * 2);
    unsigned short* wm   = (unsigned short*)carve(98304ull   * 2);
    unsigned short* wf1  = (unsigned short*)carve(49152ull   * 2);
    unsigned short* wf2  = (unsigned short*)carve(49152ull   * 2);
    unsigned short* qkvb = (unsigned short*)carve(100663296ull * 2);    // q | k | vT
    unsigned short* qb   = qkvb;
    unsigned short* kb   = qkvb + 33554432ull;
    unsigned short* vb   = qkvb + 67108864ull;                          // [bh][d][j]
    float*          qn   = (float*)carve(32768ull * 4);
    float*          kn   = (float*)carve(32768ull * 4);
    unsigned short* attn = (unsigned short*)carve(8388608ull  * 2);
    unsigned short* out2 = (unsigned short*)carve(512ull*512*PLT * 2);  // AV, padded
    float*          x2   = (float*)carve(4194304ull * 4);               // x + merge
    unsigned short* xn2  = (unsigned short*)carve(512ull*64*PLT * 2);   // GN2, padded
    unsigned short* hbuf = (unsigned short*)carve(512ull*256*PLT * 2);  // swish, padded

    // 1) pack weights to bf16 ((Cout, Cin*3) == im2col K order)
    cvt_bf16_k<<<144, 256, 0, stream>>>(w_qkv, wq, 36864);
    cvt_bf16_k<<<48,  256, 0, stream>>>(w_mrg, wm, 12288);
    cvt_bf16_k<<<24,  256, 0, stream>>>(w_ff1, wf1, 6144);
    cvt_bf16_k<<<24,  256, 0, stream>>>(w_ff2, wf2, 6144);

    // 2) GroupNorm1 -> padded rows
    groupnorm_k<<<4096, 256, 0, stream>>>(x, gn1_g, gn1_b, xn);

    // 3) QKV conv as GEMM (M=1536,N=128,K=192) per sample -> q,k,vT
    {
        GemmP p{}; p.A = wq; p.strideA = 0; p.lda = 192;
        p.B = xn; p.strideB = (size_t)64 * PLT; p.K = 192;
        p.bias = b_qkv; p.outb = qkvb;
        gemm_wmma_k<B_IM2COL, E_QKV><<<dim3(1, 24, 512), 256, 0, stream>>>(p);
    }

    // 4) row norms for cdist
    rownorm_k<<<32768, 256, 0, stream>>>(qb, qn);
    rownorm_k<<<32768, 256, 0, stream>>>(kb, kn);

    // 5) attn = 1/dist : q.k^T (M=N=256, K=1024) per head-batch
    {
        GemmP p{}; p.A = qb; p.strideA = 262144; p.lda = 1024;
        p.B = kb; p.strideB = 262144; p.ldb = 1024; p.K = 1024;
        p.qn = qn; p.kn = kn; p.outb = attn;
        gemm_wmma_k<B_TRANS, E_ATTN><<<dim3(2, 4, 128), 256, 0, stream>>>(p);
    }

    // 6) out = attn @ v (M=256,N=1024,K=256) per head-batch, B = vT
    {
        GemmP p{}; p.A = attn; p.strideA = 65536; p.lda = 256;
        p.B = vb; p.strideB = 262144; p.ldb = 256; p.K = 256;
        p.outb = out2;
        gemm_wmma_k<B_TRANS, E_AV><<<dim3(8, 4, 128), 256, 0, stream>>>(p);
    }

    // 7) merge conv + residual: x2 = x + conv(out2) (M=64,N=128,K=1536)
    {
        GemmP p{}; p.A = wm; p.strideA = 0; p.lda = 1536;
        p.B = out2; p.strideB = (size_t)512 * PLT; p.K = 1536;
        p.bias = b_mrg; p.resid = x; p.outf = x2;
        gemm_wmma_k<B_IM2COL, E_MERGE><<<dim3(1, 1, 512), 256, 0, stream>>>(p);
    }

    // 8) GroupNorm2 -> padded rows
    groupnorm_k<<<4096, 256, 0, stream>>>(x2, gn2_g, gn2_b, xn2);

    // 9) FF1 conv + swish (M=256,N=128,K=192) -> padded rows
    {
        GemmP p{}; p.A = wf1; p.strideA = 0; p.lda = 192;
        p.B = xn2; p.strideB = (size_t)64 * PLT; p.K = 192;
        p.bias = b_ff1; p.outb = hbuf;
        gemm_wmma_k<B_IM2COL, E_FF1><<<dim3(1, 4, 512), 256, 0, stream>>>(p);
    }

    // 10) FF2 conv + residual -> d_out (M=64,N=128,K=768)
    {
        GemmP p{}; p.A = wf2; p.strideA = 0; p.lda = 768;
        p.B = hbuf; p.strideB = (size_t)256 * PLT; p.K = 768;
        p.bias = b_ff2; p.resid = x2; p.outf = out;
        gemm_wmma_k<B_IM2COL, E_FF2><<<dim3(1, 1, 512), 256, 0, stream>>>(p);
    }
}
